// Patient2Vec1_26353919328295
// MI455X (gfx1250) — matrix operand order repaired and checked by reading, not compiled, not run on hardware
//
#include <hip/hip_runtime.h>
#include <hip/hip_bf16.h>

// ---- model dims ----
#define B_   32
#define S_   128
#define P_   64
#define D_   256
#define H_   128
#define G3_  384      // 3*H
#define NF_  3
#define OUT_ 2
#define LIN_IN_ 771   // 2H*NF + 3

typedef __attribute__((ext_vector_type(16))) _Float16 v16h;
typedef __attribute__((ext_vector_type(8)))  float    v8f;

// -----------------------------------------------------------------------------
// Kernel 1: fused attention pooling over pad dim.
// One block per (b,s) tile [64,256].  scores -> softmax(P) -> weighted sum.
// Single pass over the 268MB input (bandwidth-bound stage).
// -----------------------------------------------------------------------------
__global__ void attn_pool_kernel(const float* __restrict__ inputs,
                                 const float* __restrict__ conv_w,
                                 const float* __restrict__ conv_b,
                                 float* __restrict__ conv_out,      // [B*S, D]
                                 float* __restrict__ conv_wts_out)  // [B*S, P]
{
    const int bs  = blockIdx.x;            // 0..4095
    const int tid = threadIdx.x;           // 0..255
    const float* tile = inputs + (size_t)bs * (P_ * D_);

    __shared__ float s_red[256];
    __shared__ float s_scores[P_];
    __shared__ float s_wts[P_];
    __shared__ float s_misc[2];

    // hint the next tile into L2 while we chew on this one
    if (tid == 0 && bs + 1 < gridDim.x)
        __builtin_prefetch(inputs + (size_t)(bs + 1) * (P_ * D_), 0, 1);

    // ---- phase 1: scores[p] = dot(tile[p,:], conv_w) + conv_b ----
    {
        const int p = tid >> 2;            // 4 threads per p-row
        const int q = tid & 3;             // each covers 64 of 256 d's
        const float4* xr = (const float4*)(tile + p * D_ + q * 64);
        const float4* wr = (const float4*)(conv_w + q * 64);
        float acc = 0.f;
        #pragma unroll
        for (int j = 0; j < 16; ++j) {
            float4 x = xr[j], w = wr[j];
            acc += x.x * w.x + x.y * w.y + x.z * w.z + x.w * w.w;
        }
        s_red[tid] = acc;
    }
    __syncthreads();
    if (tid < P_) {
        s_scores[tid] = s_red[4 * tid] + s_red[4 * tid + 1] +
                        s_red[4 * tid + 2] + s_red[4 * tid + 3] + conv_b[0];
    }
    __syncthreads();
    // ---- softmax over P=64 ----
    if (tid == 0) {
        float m = s_scores[0];
        #pragma unroll
        for (int p = 1; p < P_; ++p) m = fmaxf(m, s_scores[p]);
        s_misc[0] = m;
    }
    __syncthreads();
    if (tid < P_) s_scores[tid] = __expf(s_scores[tid] - s_misc[0]);
    __syncthreads();
    if (tid == 0) {
        float s = 0.f;
        #pragma unroll
        for (int p = 0; p < P_; ++p) s += s_scores[p];
        s_misc[1] = 1.f / s;
    }
    __syncthreads();
    if (tid < P_) {
        float w = s_scores[tid] * s_misc[1];
        s_wts[tid] = w;
        conv_wts_out[(size_t)bs * P_ + tid] = w;
    }
    __syncthreads();
    // ---- phase 2: out[d] = sum_p w[p] * tile[p,d]  (coalesced re-read, L2-hot) ----
    {
        const int d = tid;                 // 256 threads == 256 d's
        float acc = 0.f;
        #pragma unroll 8
        for (int p = 0; p < P_; ++p)
            acc += s_wts[p] * tile[p * D_ + d];
        conv_out[(size_t)bs * D_ + d] = acc;
    }
}

// -----------------------------------------------------------------------------
// Kernel 2: f32 -> f16 conversion (grid-stride)
// -----------------------------------------------------------------------------
__global__ void cvt_f16_kernel(const float* __restrict__ src,
                               _Float16* __restrict__ dst, int n)
{
    for (int i = blockIdx.x * blockDim.x + threadIdx.x; i < n;
         i += gridDim.x * blockDim.x)
        dst[i] = (_Float16)src[i];
}

// -----------------------------------------------------------------------------
// Kernel 3: xg GEMM with WMMA.  xg[dir][m][n] = conv[m,:]·W_ih[dir][n,:] + b_ih[n]
// M=4096, N=384, K=256.  One 16x16 output tile per wave, K-loop of 8 x (16x16x32).
// Fragment layouts per CDNA5 ISA §7.12.2 (wave32).
// -----------------------------------------------------------------------------
__global__ void xg_gemm_wmma_kernel(const _Float16* __restrict__ Ah,   // [4096,256]
                                    const _Float16* __restrict__ Wh,   // [2,384,256]
                                    const float* __restrict__ b_ih_f,
                                    const float* __restrict__ b_ih_b,
                                    float* __restrict__ xg)            // [2,4096,384]
{
    const int dir   = blockIdx.y;
    const int wave  = threadIdx.x >> 5;
    const int tile  = blockIdx.x * 8 + wave;       // 0..6143  (256 * 24 tiles)
    const int mtile = tile / 24;
    const int ntile = tile % 24;

    const int lane = threadIdx.x & 31;
    const int lg   = lane >> 4;                    // lane group 0/1
    const int l16  = lane & 15;

    const _Float16* Ap = Ah + (size_t)(mtile * 16) * D_;
    const _Float16* Wp = Wh + ((size_t)dir * G3_ + ntile * 16) * D_;
    const float*    bv = dir ? b_ih_b : b_ih_f;

    v8f acc = {};
    #pragma unroll
    for (int kk = 0; kk < D_; kk += 32) {
        // A fragment: 16x32 f16.  lane<16: row=l16, K {0..7}+{16..23};
        //             lane>=16:  row=l16, K {8..15}+{24..31}
        v16h a;
        #pragma unroll
        for (int i = 0; i < 8; ++i) {
            int k0 = ((i < 4) ? (2 * i) : (16 + 2 * (i - 4))) + lg * 8;
            a[2 * i]     = Ap[l16 * D_ + kk + k0];
            a[2 * i + 1] = Ap[l16 * D_ + kk + k0 + 1];
        }
        // B fragment: 32x16 f16 (B[k][n] = W[n][k]).  lane<16: N=l16, K {0..7}+{16..23};
        //             lane>=16: N=l16, K {8..15}+{24..31} (2 consecutive K per VGPR)
        v16h b;
        #pragma unroll
        for (int i = 0; i < 8; ++i) {
            int k0 = ((i < 4) ? (2 * i) : (16 + 2 * (i - 4))) + lg * 8;
            b[2 * i]     = Wp[l16 * D_ + kk + k0];
            b[2 * i + 1] = Wp[l16 * D_ + kk + k0 + 1];
        }
        acc = __builtin_amdgcn_wmma_f32_16x16x32_f16(
            /*neg_a=*/false, a, /*neg_b=*/false, b,
            /*c_mod=*/(short)0, acc, /*reuse_a=*/false, /*reuse_b=*/false);
    }

    // C/D layout: lanes 0-15: VGPR r -> M=r; lanes 16-31: VGPR r -> M=8+r; N=l16
    const int ncol = ntile * 16 + l16;
    const float bias = bv[ncol];
    #pragma unroll
    for (int r = 0; r < 8; ++r) {
        int mrow = mtile * 16 + r + lg * 8;
        xg[((size_t)dir * (B_ * S_) + mrow) * G3_ + ncol] = acc[r] + bias;
    }
}

// -----------------------------------------------------------------------------
// Kernel 4: GRU recurrence.  One block per (dir, batch); sequential over S.
// f32 VALU (precision across 128 recurrent steps), h kept in LDS,
// W_hh (196KB) streamed from L2 with float4 loads.
// -----------------------------------------------------------------------------
__global__ __launch_bounds__(G3_)
void gru_kernel(const float* __restrict__ xg,        // [2,B*S,384]
                const float* __restrict__ W_hh_f,    // [384,128]
                const float* __restrict__ b_hh_f,
                const float* __restrict__ W_hh_b,
                const float* __restrict__ b_hh_b,
                float* __restrict__ states)          // [B,S,2H]
{
    const int dir = blockIdx.x >> 5;   // 0 fwd, 1 bwd
    const int b   = blockIdx.x & 31;
    const int tid = threadIdx.x;       // 0..383

    const float* Whh = dir ? W_hh_b : W_hh_f;
    const float* bhh = dir ? b_hh_b : b_hh_f;

    __shared__ __align__(16) float h[H_];
    __shared__ float hg[G3_];

    if (tid < H_) h[tid] = 0.f;

    const float4* wrow = (const float4*)(Whh + tid * H_);
    const float   bg   = bhh[tid];

    for (int step = 0; step < S_; ++step) {
        const int t = dir ? (S_ - 1 - step) : step;
        __syncthreads();                           // h is stable
        // hg[g] = b_hh[g] + W_hh[g,:] · h
        {
            const float4* h4 = (const float4*)h;
            float acc = bg;
            #pragma unroll 8
            for (int k = 0; k < H_ / 4; ++k) {
                float4 w = wrow[k], hv = h4[k];
                acc += w.x * hv.x + w.y * hv.y + w.z * hv.z + w.w * hv.w;
            }
            hg[tid] = acc;
        }
        __syncthreads();                           // hg ready
        if (tid < H_) {
            const size_t base = ((size_t)dir * (B_ * S_) + b * S_ + t) * G3_;
            const float xr = xg[base + tid];
            const float xz = xg[base + H_ + tid];
            const float xn = xg[base + 2 * H_ + tid];
            const float r = 1.f / (1.f + __expf(-(xr + hg[tid])));
            const float z = 1.f / (1.f + __expf(-(xz + hg[H_ + tid])));
            const float n = tanhf(xn + r * hg[2 * H_ + tid]);
            const float hn = (1.f - z) * n + z * h[tid];
            h[tid] = hn;                            // next iter guarded by top barrier
            states[((size_t)b * S_ + t) * (2 * H_) + dir * H_ + tid] = hn;
        }
    }
}

// -----------------------------------------------------------------------------
// Kernel 5: second attention + linear head + softmax.  One block per batch elem.
// -----------------------------------------------------------------------------
__global__ void attn2_head_kernel(const float* __restrict__ states,    // [B,S,2H]
                                  const float* __restrict__ conv2_w,   // [NF,2H]
                                  const float* __restrict__ conv2_b,   // [NF]
                                  const float* __restrict__ demoip,    // [B,3]
                                  const float* __restrict__ lin_w,     // [OUT,771]
                                  const float* __restrict__ lin_b,     // [OUT]
                                  float* __restrict__ alpha_out,       // [B,NF,S]
                                  float* __restrict__ context_out,     // [B,NF*2H]
                                  float* __restrict__ out_out)         // [B,OUT]
{
    const int b   = blockIdx.x;
    const int tid = threadIdx.x;       // 0..255
    const float* st = states + (size_t)b * S_ * (2 * H_);

    __shared__ float sc[NF_ * S_];
    __shared__ float alpha_s[NF_ * S_];
    __shared__ float ctx[NF_ * 2 * H_];
    __shared__ float logits[OUT_];

    // alpha_scores[f,s] = states[b,s,:] · conv2_w[f,:] + conv2_b[f]
    for (int idx = tid; idx < NF_ * S_; idx += blockDim.x) {
        const int f = idx / S_, s = idx % S_;
        const float4* xr = (const float4*)(st + s * (2 * H_));
        const float4* wr = (const float4*)(conv2_w + f * (2 * H_));
        float acc = conv2_b[f];
        #pragma unroll 8
        for (int j = 0; j < (2 * H_) / 4; ++j) {
            float4 x = xr[j], w = wr[j];
            acc += x.x * w.x + x.y * w.y + x.z * w.z + x.w * w.w;
        }
        sc[idx] = acc;
    }
    __syncthreads();
    // softmax over s per filter
    if (tid < NF_) {
        const int f = tid;
        float m = sc[f * S_];
        for (int s = 1; s < S_; ++s) m = fmaxf(m, sc[f * S_ + s]);
        float sum = 0.f;
        for (int s = 0; s < S_; ++s) sum += __expf(sc[f * S_ + s] - m);
        const float inv = 1.f / sum;
        for (int s = 0; s < S_; ++s) {
            float a = __expf(sc[f * S_ + s] - m) * inv;
            alpha_s[f * S_ + s] = a;
            alpha_out[((size_t)b * NF_ + f) * S_ + s] = a;
        }
    }
    __syncthreads();
    // context[f,hh] = sum_s alpha[f,s] * states[b,s,hh]
    for (int idx = tid; idx < NF_ * 2 * H_; idx += blockDim.x) {
        const int f = idx / (2 * H_), hh = idx % (2 * H_);
        float acc = 0.f;
        #pragma unroll 8
        for (int s = 0; s < S_; ++s)
            acc += alpha_s[f * S_ + s] * st[s * (2 * H_) + hh];
        ctx[idx] = acc;
        context_out[(size_t)b * (NF_ * 2 * H_) + idx] = acc;
    }
    __syncthreads();
    // linear + softmax over OUT=2
    if (tid < OUT_) {
        const int o = tid;
        const float* w = lin_w + o * LIN_IN_;
        float acc = lin_b[o];
        for (int i = 0; i < NF_ * 2 * H_; ++i) acc += w[i] * ctx[i];
        for (int j = 0; j < 3; ++j) acc += w[NF_ * 2 * H_ + j] * demoip[b * 3 + j];
        logits[o] = acc;
    }
    __syncthreads();
    if (tid < OUT_) {
        const float m  = fmaxf(logits[0], logits[1]);
        const float e0 = __expf(logits[0] - m), e1 = __expf(logits[1] - m);
        out_out[b * OUT_ + tid] = __expf(logits[tid] - m) / (e0 + e1);
    }
}

// -----------------------------------------------------------------------------
extern "C" void kernel_launch(void* const* d_in, const int* in_sizes, int n_in,
                              void* d_out, int out_size, void* d_ws, size_t ws_size,
                              hipStream_t stream) {
    const float* inputs   = (const float*)d_in[0];
    const float* demoip   = (const float*)d_in[1];
    const float* conv_w   = (const float*)d_in[2];
    const float* conv_b   = (const float*)d_in[3];
    const float* conv2_w  = (const float*)d_in[4];
    const float* conv2_b  = (const float*)d_in[5];
    const float* W_ih_f   = (const float*)d_in[6];
    const float* W_hh_f   = (const float*)d_in[7];
    const float* b_ih_f   = (const float*)d_in[8];
    const float* b_hh_f   = (const float*)d_in[9];
    const float* W_ih_b   = (const float*)d_in[10];
    const float* W_hh_b   = (const float*)d_in[11];
    const float* b_ih_b   = (const float*)d_in[12];
    const float* b_hh_b   = (const float*)d_in[13];
    const float* lin_w    = (const float*)d_in[14];
    const float* lin_b    = (const float*)d_in[15];

    // output layout (flat, return order)
    float* out_f = (float*)d_out;
    float* out_out     = out_f;                         // 64
    float* alpha_out   = out_f + 64;                    // 12288
    float* states_out  = out_f + 64 + 12288;            // 1048576
    float* context_out = out_f + 64 + 12288 + 1048576;  // 24576
    float* convwts_out = out_f + 64 + 12288 + 1048576 + 24576; // 262144

    // workspace layout
    float*    conv  = (float*)d_ws;                         // [4096,256]
    float*    xg    = conv + (size_t)B_ * S_ * D_;          // [2,4096,384]
    _Float16* convh = (_Float16*)(xg + (size_t)2 * B_ * S_ * G3_);
    _Float16* wih16 = convh + (size_t)B_ * S_ * D_;         // [2,384,256]

    // 1) fused attention pooling (one pass over 268MB)
    attn_pool_kernel<<<B_ * S_, 256, 0, stream>>>(
        inputs, conv_w, conv_b, conv, convwts_out);

    // 2) f32->f16 for the WMMA GEMM
    cvt_f16_kernel<<<1024, 256, 0, stream>>>(conv, convh, B_ * S_ * D_);
    cvt_f16_kernel<<<96,   256, 0, stream>>>(W_ih_f, wih16, G3_ * D_);
    cvt_f16_kernel<<<96,   256, 0, stream>>>(W_ih_b, wih16 + (size_t)G3_ * D_, G3_ * D_);

    // 3) xg = conv @ W_ih^T + b_ih, both directions, via v_wmma_f32_16x16x32_f16
    //    6144 tiles/dir, 8 waves (tiles) per block
    xg_gemm_wmma_kernel<<<dim3(768, 2), 256, 0, stream>>>(
        convh, wih16, b_ih_f, b_ih_b, xg);

    // 4) GRU recurrence: 64 independent (dir,batch) chains
    gru_kernel<<<64, G3_, 0, stream>>>(
        xg, W_hh_f, b_hh_f, W_hh_b, b_hh_b, states_out);

    // 5) second attention + linear head
    attn2_head_kernel<<<B_, 256, 0, stream>>>(
        states_out, conv2_w, conv2_b, demoip, lin_w, lin_b,
        alpha_out, context_out, out_out);
}
